// DeformBasicBlock1_44195213476554
// MI455X (gfx1250) — matrix operand order repaired and checked by reading, not compiled, not run on hardware
//
#include <hip/hip_runtime.h>
#include <hip/hip_bf16.h>
#include <math.h>
#include <stdint.h>

typedef __attribute__((ext_vector_type(16))) _Float16 v16h;
typedef __attribute__((ext_vector_type(8)))  float    v8f;
typedef unsigned int u32x4 __attribute__((ext_vector_type(4)));
typedef unsigned int u32x8 __attribute__((ext_vector_type(8)));

// Problem constants (from setup_inputs)
constexpr int Bn = 2, Cn = 64, Dn = 8, Hn = 56, Wn = 56;
constexpr int SP = Dn * Hn * Wn;          // 25088
constexpr int Mtot = Bn * SP;             // 50176
constexpr int KT = Cn * 27;               // 1728
constexpr int OFFC = 648;                 // G*3*K
constexpr int KK = 27;

__device__ __forceinline__ void decode_m(int m, int& b_, int& d_, int& h_, int& w_) {
  w_ = m % Wn; int t = m / Wn;
  h_ = t % Hn; t /= Hn;
  d_ = t % Dn; b_ = t / Dn;
}

// ---------------------------------------------------------------------------
// WMMA fragment loaders (wave32); A and B share this code's contiguous k order.
// A: 16x32 f16 row-major (stride 32).  lane<16 -> K 0..15, lane>=16 -> K 16..31;
//    element j: K=base+j. m = lane&15.
// B stored [n][k] (stride 32): lane gives N=(lane&15)+noff, K run =
//    (lane<16 ? 0..15 : 16..31) -> one contiguous 16-half read per lane.
// C/D f32 16x16: vgpr r at lane L -> M = r + (L<16?0:8), N = L&15.
// ---------------------------------------------------------------------------
__device__ __forceinline__ v16h load_frag_A(const _Float16* lds, int lane) {
  int m = lane & 15;
  int kbase = (lane < 16) ? 0 : 16;
  v16h a;
#pragma unroll
  for (int j = 0; j < 16; ++j) {
    a[j] = lds[m * 32 + kbase + j];
  }
  return a;
}

__device__ __forceinline__ v16h load_frag_B_nk(const _Float16* lds, int noff, int lane) {
  int n = (lane & 15) + noff;
  const _Float16* p = lds + n * 32 + ((lane < 16) ? 0 : 16);
  v16h b;
#pragma unroll
  for (int j = 0; j < 16; ++j) b[j] = p[j];
  return b;
}

// ---------------------------------------------------------------------------
// Tensor Data Mover: DMA a [tile1 rows x 32 halves] weight tile (row stride
// stride0 halves) from global into LDS at lds_addr, row-major [n][k].
// D# per CDNA5 ISA: group0 = count/lds/global/type,
// group1 = data_size(2B), tensor dims (for edge zero-fill), tile dims, strides.
// Issued by one wave; completion via s_wait_tensorcnt.
// ---------------------------------------------------------------------------
__device__ __forceinline__ void tdm_load_tile(const _Float16* gsrc, unsigned lds_addr,
                                              unsigned tdim0, unsigned tdim1,
                                              unsigned tile1, unsigned stride0) {
  unsigned long long ga = (unsigned long long)(uintptr_t)gsrc;
  u32x4 g0;
  g0[0] = 1u;                                   // count=1, user descriptor
  g0[1] = lds_addr;                             // LDS byte address
  g0[2] = (unsigned)ga;                         // global addr [31:0]
  g0[3] = (unsigned)((ga >> 32) & 0x01ffffffull) | 0x80000000u;  // [56:32] | type=2
  u32x8 g1;
  g1[0] = 0x00010000u;                          // workgroup_mask=0, data_size=2B
  g1[1] = (tdim0 & 0xffffu) << 16;              // tensor_dim0 lo
  g1[2] = (tdim0 >> 16) | ((tdim1 & 0xffffu) << 16);  // tensor_dim0 hi | dim1 lo
  g1[3] = (tdim1 >> 16) | (32u << 16);          // tensor_dim1 hi | tile_dim0=32
  g1[4] = tile1;                                // tile_dim1 | tile_dim2=0
  g1[5] = stride0;                              // tensor_dim0_stride lo
  g1[6] = 0u;
  g1[7] = 0u;
  asm volatile("tensor_load_to_lds %0, %1" : : "s"(g0), "s"(g1) : "memory");
}

// ---------------------------------------------------------------------------
__global__ void cvt_f16_kernel(const float* __restrict__ in, _Float16* __restrict__ out, int n) {
  int i = blockIdx.x * blockDim.x + threadIdx.x;
  if (i < n) out[i] = (_Float16)in[i];
}

// ---------------------------------------------------------------------------
// Implicit-GEMM 3x3x3 SAME conv. Block tile 32M x 128N, 4 waves; wave w owns
// (mt = w>>1)*16 rows x (nt = w&1)*64 cols as 4 WMMA accumulators.
// ---------------------------------------------------------------------------
__global__ void conv3d_wmma_kernel(const _Float16* __restrict__ xh,
                                   const _Float16* __restrict__ wgt,  // [Nc][KT]
                                   const float* __restrict__ bias,
                                   float* __restrict__ out, int Nc) {
  __shared__ _Float16 As[32 * 32];
  __shared__ _Float16 Bs[128 * 32];
  const int mtile = blockIdx.x * 32;
  const int ntile0 = blockIdx.y * 128;
  const int tid = threadIdx.x;
  const int wave = tid >> 5;
  const int lane = tid & 31;
  const int mt = wave >> 1;
  const int nt = wave & 1;

  // rows this thread stages: ml = wave + 4p, column kk = lane (constant)
  int mdec[8];
#pragma unroll
  for (int p = 0; p < 8; ++p) {
    int b_, d_, h_, w_;
    decode_m(mtile + wave + 4 * p, b_, d_, h_, w_);
    mdec[p] = (((b_ * 8 + d_) * 64 + h_) * 64) + w_;
  }
  const int kk = lane;
  int c = kk / 27;                // incremental base-27 counter thereafter
  int r = kk - 27 * c;

  const unsigned ldsB = (unsigned)(uintptr_t)(void*)Bs;
  v8f acc[4] = {};

  for (int k0 = 0; k0 < KT; k0 += 32) {
    if (wave == 0) {
      tdm_load_tile(wgt + (size_t)ntile0 * KT + k0, ldsB,
                    (unsigned)(KT - k0), (unsigned)(Nc - ntile0), 128u, (unsigned)KT);
      __builtin_amdgcn_s_wait_tensorcnt(0);
    }
    // ---- stage A tile (im2col): decode (c, kd, kh, kw) once per K-step ----
    const int dd = r / 9 - 1, hh = (r / 3) % 3 - 1, ww = r % 3 - 1;
#pragma unroll
    for (int p = 0; p < 8; ++p) {
      int enc = mdec[p];
      int w_ = enc & 63, h_ = (enc >> 6) & 63, d_ = (enc >> 12) & 7, b_ = enc >> 15;
      int zd = d_ + dd, zh = h_ + hh, zw = w_ + ww;
      _Float16 v = (_Float16)0.f;
      if (zd >= 0 && zd < Dn && zh >= 0 && zh < Hn && zw >= 0 && zw < Wn)
        v = xh[(((b_ * Cn + c) * Dn + zd) * Hn + zh) * Wn + zw];
      As[(wave + 4 * p) * 32 + kk] = v;
    }
    __syncthreads();
    v16h a = load_frag_A(As + mt * 16 * 32, lane);
#pragma unroll
    for (int s = 0; s < 4; ++s) {
      v16h b = load_frag_B_nk(Bs, nt * 64 + s * 16, lane);
      acc[s] = __builtin_amdgcn_wmma_f32_16x16x32_f16(false, a, false, b, (short)0,
                                                      acc[s], false, false);
    }
    __syncthreads();
    r += 32; if (r >= 27) { r -= 27; ++c; } if (r >= 27) { r -= 27; ++c; }
  }

  const int n0 = ntile0 + nt * 64;
#pragma unroll
  for (int s = 0; s < 4; ++s) {
#pragma unroll
    for (int rr = 0; rr < 8; ++rr) {
      int ml = rr + ((lane < 16) ? 0 : 8);
      int n = n0 + s * 16 + (lane & 15);
      if (n < Nc) {
        int b_, d_, h_, w_;
        decode_m(mtile + mt * 16 + ml, b_, d_, h_, w_);
        out[(((b_ * Nc + n) * Dn + d_) * Hn + h_) * Wn + w_] = acc[s][rr] + bias[n];
      }
    }
  }
}

// ---------------------------------------------------------------------------
// Deformable conv as GEMM: block tile 64M x 64N, 4 waves; wave w owns rows
// w*16..+16 and all 64 cols (4 accumulators). A elements are trilinear
// gathers steered by the offset tensor; B = w_dc flat [64][1728] via TDM.
// ---------------------------------------------------------------------------
__global__ void deform_wmma_kernel(const float* __restrict__ x,
                                   const float* __restrict__ off,    // [B,648,D,H,W]
                                   const _Float16* __restrict__ wgt, // [64][KT]
                                   float* __restrict__ out) {
  __shared__ _Float16 As[64 * 32];
  __shared__ _Float16 Bs[64 * 32];
  const int mtile = blockIdx.x * 64;
  const int tid = threadIdx.x;
  const int wave = tid >> 5;
  const int lane = tid & 31;

  int mdec[16];
#pragma unroll
  for (int p = 0; p < 16; ++p) {
    int b_, d_, h_, w_;
    decode_m(mtile + wave + 4 * p, b_, d_, h_, w_);
    mdec[p] = (((b_ * 8 + d_) * 64 + h_) * 64) + w_;
  }
  const int kk = lane;
  int cf = kk / 27;               // full channel index (g*8 + c)
  int k = kk - 27 * cf;           // kernel tap 0..26

  const unsigned ldsB = (unsigned)(uintptr_t)(void*)Bs;
  v8f acc[4] = {};

  for (int k0 = 0; k0 < KT; k0 += 32) {
    if (wave == 0) {
      tdm_load_tile(wgt + k0, ldsB, (unsigned)(KT - k0), 64u, 64u, (unsigned)KT);
      __builtin_amdgcn_s_wait_tensorcnt(0);
    }
    // per-K-step decode (shared across the 16 rows this thread stages)
    const int g = cf >> 3;
    const int ch = (g * KK + k) * 3;
    const float kd = (float)(k / 9 - 1);
    const float kh = (float)((k / 3) % 3 - 1);
    const float kw = (float)(k % 3 - 1);
#pragma unroll
    for (int p = 0; p < 16; ++p) {
      int enc = mdec[p];
      int w_ = enc & 63, h_ = (enc >> 6) & 63, d_ = (enc >> 12) & 7, b_ = enc >> 15;
      const float* ob = off + ((b_ * OFFC + ch) * SP) + (d_ * Hn + h_) * Wn + w_;
      float pd = ob[0]      + kd + (float)d_;
      float ph = ob[SP]     + kh + (float)h_;
      float pw = ob[2 * SP] + kw + (float)w_;
      const float* xc = x + (b_ * Cn + cf) * SP;
      float fd0 = floorf(pd), fh0 = floorf(ph), fw0 = floorf(pw);
      float fd = pd - fd0, fh = ph - fh0, fw = pw - fw0;
      float s = 0.f;
#pragma unroll
      for (int t2 = 0; t2 < 8; ++t2) {
        int ddi = t2 >> 2, hhi = (t2 >> 1) & 1, wwi = t2 & 1;
        float dc = fd0 + ddi, hc = fh0 + hhi, wc = fw0 + wwi;
        float wt = (ddi ? fd : 1.f - fd) * (hhi ? fh : 1.f - fh) * (wwi ? fw : 1.f - fw);
        bool valid = (dc >= 0.f) & (dc < (float)Dn) & (hc >= 0.f) &
                     (hc < (float)Hn) & (wc >= 0.f) & (wc < (float)Wn);
        int di = (int)fminf(fmaxf(dc, 0.f), (float)(Dn - 1));
        int hi = (int)fminf(fmaxf(hc, 0.f), (float)(Hn - 1));
        int wi = (int)fminf(fmaxf(wc, 0.f), (float)(Wn - 1));
        float v = xc[(di * Hn + hi) * Wn + wi];
        s += v * (valid ? wt : 0.f);
      }
      As[(wave + 4 * p) * 32 + kk] = (_Float16)s;
    }
    __syncthreads();
    v16h a = load_frag_A(As + wave * 16 * 32, lane);
#pragma unroll
    for (int s = 0; s < 4; ++s) {
      v16h b = load_frag_B_nk(Bs, s * 16, lane);
      acc[s] = __builtin_amdgcn_wmma_f32_16x16x32_f16(false, a, false, b, (short)0,
                                                      acc[s], false, false);
    }
    __syncthreads();
    k += 32; if (k >= 27) { k -= 27; ++cf; } if (k >= 27) { k -= 27; ++cf; }
  }

#pragma unroll
  for (int s = 0; s < 4; ++s) {
#pragma unroll
    for (int rr = 0; rr < 8; ++rr) {
      int ml = rr + ((lane < 16) ? 0 : 8);
      int n = s * 16 + (lane & 15);
      int b_, d_, h_, w_;
      decode_m(mtile + wave * 16 + ml, b_, d_, h_, w_);
      out[(((b_ * Cn + n) * Dn + d_) * Hn + h_) * Wn + w_] = acc[s][rr];
    }
  }
}

// ---------------------------------------------------------------------------
// BatchNorm stats: one block per channel.
// ---------------------------------------------------------------------------
__global__ void bn_stats_kernel(const float* __restrict__ x,
                                const float* __restrict__ gamma,
                                const float* __restrict__ beta,
                                float* __restrict__ scale,
                                float* __restrict__ shift) {
  __shared__ float ssum[256];
  __shared__ float ssq[256];
  const int c = blockIdx.x;
  float s = 0.f, q = 0.f;
  for (int i = threadIdx.x; i < Bn * SP; i += 256) {
    int b = i / SP, r = i % SP;
    float v = x[(b * Cn + c) * SP + r];
    s += v; q += v * v;
  }
  ssum[threadIdx.x] = s; ssq[threadIdx.x] = q;
  __syncthreads();
  for (int st = 128; st > 0; st >>= 1) {
    if (threadIdx.x < st) {
      ssum[threadIdx.x] += ssum[threadIdx.x + st];
      ssq[threadIdx.x]  += ssq[threadIdx.x + st];
    }
    __syncthreads();
  }
  if (threadIdx.x == 0) {
    const float inv = 1.f / (float)(Bn * SP);
    float mean = ssum[0] * inv;
    float var = ssq[0] * inv - mean * mean;
    float sc = gamma[c] * rsqrtf(var + 1e-5f);
    scale[c] = sc;
    shift[c] = beta[c] - mean * sc;
  }
}

__global__ void bn_relu_kernel(const float* __restrict__ in,
                               const float* __restrict__ scale,
                               const float* __restrict__ shift,
                               float* __restrict__ out32,
                               _Float16* __restrict__ out16) {
  int i = blockIdx.x * blockDim.x + threadIdx.x;
  if (i >= Bn * Cn * SP) return;
  int c = (i / SP) % Cn;
  float v = fmaxf(in[i] * scale[c] + shift[c], 0.f);
  out32[i] = v;
  out16[i] = (_Float16)v;
}

__global__ void bn_add_relu_kernel(const float* __restrict__ in,
                                   const float* __restrict__ scale,
                                   const float* __restrict__ shift,
                                   const float* __restrict__ residual,
                                   float* __restrict__ out) {
  int i = blockIdx.x * blockDim.x + threadIdx.x;
  if (i >= Bn * Cn * SP) return;
  int c = (i / SP) % Cn;
  out[i] = fmaxf(in[i] * scale[c] + shift[c] + residual[i], 0.f);
}

// ---------------------------------------------------------------------------
extern "C" void kernel_launch(void* const* d_in, const int* in_sizes, int n_in,
                              void* d_out, int out_size, void* d_ws, size_t ws_size,
                              hipStream_t stream) {
  const float* x      = (const float*)d_in[0];
  const float* w_off1 = (const float*)d_in[1];
  const float* b_off1 = (const float*)d_in[2];
  const float* w_dc1  = (const float*)d_in[3];
  const float* gamma1 = (const float*)d_in[4];
  const float* beta1  = (const float*)d_in[5];
  const float* w_off2 = (const float*)d_in[6];
  const float* b_off2 = (const float*)d_in[7];
  const float* w_dc2  = (const float*)d_in[8];
  const float* gamma2 = (const float*)d_in[9];
  const float* beta2  = (const float*)d_in[10];
  float* out = (float*)d_out;

  char* ws = (char*)d_ws;
  size_t cur = 0;
  auto alloc = [&](size_t bytes) -> void* {
    size_t p = (cur + 255) & ~(size_t)255;
    cur = p + bytes;
    return (void*)(ws + p);
  };

  const size_t NX = (size_t)Bn * Cn * SP;
  const size_t NWOFF = (size_t)OFFC * KT;
  const size_t NWDC = (size_t)Cn * KT;

  _Float16* xh     = (_Float16*)alloc(NX * 2);
  _Float16* woff1h = (_Float16*)alloc(NWOFF * 2);
  _Float16* wdc1h  = (_Float16*)alloc(NWDC * 2);
  _Float16* woff2h = (_Float16*)alloc(NWOFF * 2);
  _Float16* wdc2h  = (_Float16*)alloc(NWDC * 2);
  float* offbuf    = (float*)alloc((size_t)Bn * OFFC * SP * 4);  // reused off1/off2
  float* hraw      = (float*)alloc(NX * 4);                      // reused h1raw/h2raw
  float* h1        = (float*)alloc(NX * 4);
  _Float16* h1h    = (_Float16*)alloc(NX * 2);
  float* scale1    = (float*)alloc(Cn * 4);
  float* shift1    = (float*)alloc(Cn * 4);
  float* scale2    = (float*)alloc(Cn * 4);
  float* shift2    = (float*)alloc(Cn * 4);

  const int CB = 256;
  auto grid1 = [&](size_t n) { return dim3((unsigned)((n + CB - 1) / CB)); };

  cvt_f16_kernel<<<grid1(NX), CB, 0, stream>>>(x, xh, (int)NX);
  cvt_f16_kernel<<<grid1(NWOFF), CB, 0, stream>>>(w_off1, woff1h, (int)NWOFF);
  cvt_f16_kernel<<<grid1(NWDC), CB, 0, stream>>>(w_dc1, wdc1h, (int)NWDC);
  cvt_f16_kernel<<<grid1(NWOFF), CB, 0, stream>>>(w_off2, woff2h, (int)NWOFF);
  cvt_f16_kernel<<<grid1(NWDC), CB, 0, stream>>>(w_dc2, wdc2h, (int)NWDC);

  const dim3 convGrid(Mtot / 32, (OFFC + 127) / 128);  // 1568 x 6
  const dim3 dcGrid(Mtot / 64, 1);                     // 784
  const dim3 gemmBlock(128);

  // ---- first half ----
  conv3d_wmma_kernel<<<convGrid, gemmBlock, 0, stream>>>(xh, woff1h, b_off1, offbuf, OFFC);
  deform_wmma_kernel<<<dcGrid, gemmBlock, 0, stream>>>(x, offbuf, wdc1h, hraw);
  bn_stats_kernel<<<Cn, 256, 0, stream>>>(hraw, gamma1, beta1, scale1, shift1);
  bn_relu_kernel<<<grid1(NX), CB, 0, stream>>>(hraw, scale1, shift1, h1, h1h);

  // ---- second half ----
  conv3d_wmma_kernel<<<convGrid, gemmBlock, 0, stream>>>(h1h, woff2h, b_off2, offbuf, OFFC);
  deform_wmma_kernel<<<dcGrid, gemmBlock, 0, stream>>>(h1, offbuf, wdc2h, hraw);
  bn_stats_kernel<<<Cn, 256, 0, stream>>>(hraw, gamma2, beta2, scale2, shift2);
  bn_add_relu_kernel<<<grid1(NX), CB, 0, stream>>>(hraw, scale2, shift2, x, out);

  (void)in_sizes; (void)n_in; (void)out_size; (void)ws_size;
}
